// GNN_89335319757549
// MI455X (gfx1250) — matrix operand order repaired and checked by reading, not compiled, not run on hardware
//
#include <hip/hip_runtime.h>
#include <math.h>

// ---------------------------------------------------------------------------
// GIN forward (4 layers) for MI455X / gfx1250.
//  - scatter-add over edges: wave32 per edge, float4 gather, native f32 atomics
//  - GEMMs: bf16 WMMA (v_wmma_f32_16x16x32_bf16), fp32 accum
//  - weights pre-converted to bf16 (transposed) once per layer, then staged
//    into LDS via GLOBAL_LOAD_ASYNC_TO_LDS_B128 (ASYNCcnt), overlapped with
//    the VALU-heavy A-tile staging
//  - BN training stats fused into GEMM1 epilogue; BN+ELU fused into GEMM2
// ---------------------------------------------------------------------------

#define N_NODES 100000
#define N_EDGES 1600000
#define DCH     128
#define NLAYERS 4
#define NGRAPH  512
#define BN_EPS  1e-5f

#define AS_STRIDE 136   // padded row stride (bf16 units): 68 dwords == 4 mod 64 banks
#define WS_STRIDE 136

typedef __attribute__((ext_vector_type(16))) __bf16          v16bf;
typedef __attribute__((ext_vector_type(16))) unsigned short  v16u;
typedef __attribute__((ext_vector_type(8)))  float           v8f;

__device__ __forceinline__ unsigned short f2bf(float f) {
  unsigned int u = __builtin_bit_cast(unsigned int, f);
  u += 0x7fffu + ((u >> 16) & 1u);          // round-to-nearest-even
  return (unsigned short)(u >> 16);
}

__device__ __forceinline__ float eluf(float x) {
  return x > 0.f ? x : (__expf(x) - 1.f);
}

__device__ __forceinline__ void gatomic_add(float* p, float v) {
  // Relaxed, agent-scope -> global_atomic_add_f32 (no CAS loop).
  __hip_atomic_fetch_add(p, v, __ATOMIC_RELAXED, __HIP_MEMORY_SCOPE_AGENT);
}

// ---- zero fill ------------------------------------------------------------
__global__ void zero4_kernel(float4* __restrict__ p, int n4) {
  int i = blockIdx.x * blockDim.x + threadIdx.x;
  if (i < n4) p[i] = make_float4(0.f, 0.f, 0.f, 0.f);
}

// ---- per-layer weight prep: bf16, transposed (wt[n*128+k] = bf16(W[k][n]))
__global__ __launch_bounds__(256)
void prep_weights(const float* __restrict__ W1l, const float* __restrict__ W2l,
                  unsigned short* __restrict__ wt1, unsigned short* __restrict__ wt2) {
  int i = blockIdx.x * blockDim.x + threadIdx.x;      // 0..8191
#pragma unroll
  for (int j = 0; j < 4; ++j) {
    int idx = i * 4 + j;                              // 0..32767
    int m = idx >> 14;                                // 0: W1, 1: W2
    int e = idx & 16383;
    int n = e >> 7, k = e & 127;
    const float* W = m ? W2l : W1l;
    unsigned short* o = m ? wt2 : wt1;
    o[n * DCH + k] = f2bf(W[k * DCH + n]);
  }
}

// ---- edge scatter-add: agg[dst] += h[src] ---------------------------------
__global__ __launch_bounds__(256)
void scatter_kernel(const float* __restrict__ h, const int* __restrict__ src,
                    const int* __restrict__ dst, float* __restrict__ agg) {
  long t = (long)blockIdx.x * 256 + threadIdx.x;
  int e = (int)(t >> 5), lane = (int)(t & 31);
  if (e >= N_EDGES) return;
  if (lane == 0 && e + 4096 < N_EDGES) {
    __builtin_prefetch(src + e + 4096, 0, 0);
    __builtin_prefetch(dst + e + 4096, 0, 0);
  }
  int s = src[e], d = dst[e];
  const float4 v = *(const float4*)(h + (size_t)s * DCH + lane * 4);
  float* o = agg + (size_t)d * DCH + lane * 4;
  gatomic_add(o + 0, v.x); gatomic_add(o + 1, v.y);
  gatomic_add(o + 2, v.z); gatomic_add(o + 3, v.w);
}

// ---- async stage of bf16 weight tile into padded LDS ----------------------
__device__ __forceinline__ void stage_weights_async(unsigned short* Wt,
                                                    const unsigned short* wtg,
                                                    int tid) {
  unsigned long long gbase = (unsigned long long)(uintptr_t)wtg;
#pragma unroll
  for (int j = 0; j < 8; ++j) {
    int idx = (tid + j * 256) * 8;                    // element index (8 bf16/chunk)
    int n = idx >> 7, k = idx & 127;
    unsigned lds  = (unsigned)(uintptr_t)(Wt + n * WS_STRIDE + k);
    unsigned voff = (unsigned)(idx * 2);              // byte offset in global
    asm volatile("global_load_async_to_lds_b128 %0, %1, %2"
                 :: "v"(lds), "v"(voff), "s"(gbase) : "memory");
  }
}

__device__ __forceinline__ void wait_async_lds() {
  asm volatile("s_wait_asynccnt 0x0" ::: "memory");
}

// ---- WMMA fragment loaders (ISA 16-bit A/B lane layouts, wave32) ----------
__device__ __forceinline__ v16bf load_a_frag(const unsigned short* ap, int k0,
                                             int half) {
  v16u au;
#pragma unroll
  for (int v = 0; v < 4; ++v) {
    unsigned int p0 = *(const unsigned int*)(ap + k0 + half * 8 + 2 * v);
    au[2 * v]     = (unsigned short)p0;
    au[2 * v + 1] = (unsigned short)(p0 >> 16);
    unsigned int p1 = *(const unsigned int*)(ap + k0 + 16 + half * 8 + 2 * v);
    au[8 + 2 * v] = (unsigned short)p1;
    au[9 + 2 * v] = (unsigned short)(p1 >> 16);
  }
  return __builtin_bit_cast(v16bf, au);
}

__device__ __forceinline__ v16bf load_b_frag(const unsigned short* bp) {
  v16u bu;
#pragma unroll
  for (int v = 0; v < 8; ++v) {
    unsigned int p = *(const unsigned int*)(bp + 2 * v + (v >= 4 ? 8 : 0));
    bu[2 * v]     = (unsigned short)p;
    bu[2 * v + 1] = (unsigned short)(p >> 16);
  }
  return __builtin_bit_cast(v16bf, bu);
}

// ---- GEMM1: z = (h+agg)@W1 + b1, accumulate BN sum/sumsq ------------------
__global__ __launch_bounds__(256)
void gin_gemm1(const float* __restrict__ hin, const float* agg,
               const unsigned short* __restrict__ wtg, const float* __restrict__ bias,
               float* zout, float* gsum, float* gsq) {
  __shared__ __align__(16) unsigned short As[64 * AS_STRIDE];
  __shared__ __align__(16) unsigned short Wt[DCH * WS_STRIDE];
  __shared__ float bs[DCH];
  __shared__ float colSum[DCH];
  __shared__ float colSq[DCH];

  const int tid  = threadIdx.x;
  const int row0 = blockIdx.x * 64;

  stage_weights_async(Wt, wtg, tid);                  // ASYNCcnt copies in flight
  if (tid < DCH) { bs[tid] = bias[tid]; colSum[tid] = 0.f; colSq[tid] = 0.f; }

  // Stage A tile: t = h + agg, bf16 (overlaps with async weight DMA).
  for (int i = tid; i < 64 * DCH; i += 256) {
    int r = i >> 7, c = i & 127;
    int row = row0 + r;
    float v = 0.f;
    if (row < N_NODES) v = hin[(size_t)row * DCH + c] + agg[(size_t)row * DCH + c];
    As[r * AS_STRIDE + c] = f2bf(v);
  }
  wait_async_lds();
  __syncthreads();

  const int lane = tid & 31, wave = tid >> 5;
  const int rt = wave & 3;            // row tile 0..3
  const int cg = wave >> 2;           // column group (4 tiles each)
  const int half = lane >> 4, lc = lane & 15;

  v8f acc[4] = {};
  const unsigned short* ap = As + (rt * 16 + lc) * AS_STRIDE;
#pragma unroll
  for (int k0 = 0; k0 < DCH; k0 += 32) {
    v16bf a = load_a_frag(ap, k0, half);
#pragma unroll
    for (int t = 0; t < 4; ++t) {
      int n = (cg * 4 + t) * 16 + lc;
      v16bf b = load_b_frag(Wt + n * WS_STRIDE + k0 + half * 8);
      acc[t] = __builtin_amdgcn_wmma_f32_16x16x32_bf16(
          false, a, false, b, (short)0, acc[t], false, false);
    }
  }

  // Epilogue: bias, store z, per-column BN partials.
#pragma unroll
  for (int t = 0; t < 4; ++t) {
    const int col = (cg * 4 + t) * 16 + lc;
    const float bv = bs[col];
    float s = 0.f, s2 = 0.f;
#pragma unroll
    for (int v = 0; v < 8; ++v) {
      int row = row0 + rt * 16 + half * 8 + v;
      if (row < N_NODES) {
        float z = acc[t][v] + bv;
        zout[(size_t)row * DCH + col] = z;
        s += z; s2 += z * z;
      }
    }
    atomicAdd(&colSum[col], s);
    atomicAdd(&colSq[col], s2);
  }
  __syncthreads();
  if (tid < DCH) {
    gatomic_add(&gsum[tid], colSum[tid]);
    gatomic_add(&gsq[tid], colSq[tid]);
  }
}

// ---- BN finalize: scale/shift from accumulated stats ----------------------
__global__ void bn_finalize(const float* __restrict__ gsum, const float* __restrict__ gsq,
                            const float* __restrict__ gamma, const float* __restrict__ beta,
                            float* __restrict__ scale, float* __restrict__ shift) {
  int c = threadIdx.x;
  if (c >= DCH) return;
  float mu  = gsum[c] * (1.f / (float)N_NODES);
  float var = gsq[c] * (1.f / (float)N_NODES) - mu * mu;
  float sc  = gamma[c] * rsqrtf(var + BN_EPS);
  scale[c] = sc;
  shift[c] = beta[c] - mu * sc;
}

// ---- GEMM2: h = elu( elu(bn(z)) @ W2 + b2 ) -------------------------------
__global__ __launch_bounds__(256)
void gin_gemm2(const float* __restrict__ zin, const float* __restrict__ scale,
               const float* __restrict__ shift, const unsigned short* __restrict__ wtg,
               const float* __restrict__ bias, float* __restrict__ hout) {
  __shared__ __align__(16) unsigned short As[64 * AS_STRIDE];
  __shared__ __align__(16) unsigned short Wt[DCH * WS_STRIDE];
  __shared__ float bs[DCH], sc[DCH], sh[DCH];

  const int tid  = threadIdx.x;
  const int row0 = blockIdx.x * 64;

  stage_weights_async(Wt, wtg, tid);                  // ASYNCcnt copies in flight
  if (tid < DCH) { bs[tid] = bias[tid]; sc[tid] = scale[tid]; sh[tid] = shift[tid]; }
  __syncthreads();

  // Stage A tile: elu(bn(z)), bf16 (overlaps with async weight DMA).
  for (int i = tid; i < 64 * DCH; i += 256) {
    int r = i >> 7, c = i & 127;
    int row = row0 + r;
    float v = 0.f;
    if (row < N_NODES) v = eluf(zin[(size_t)row * DCH + c] * sc[c] + sh[c]);
    As[r * AS_STRIDE + c] = f2bf(v);
  }
  wait_async_lds();
  __syncthreads();

  const int lane = tid & 31, wave = tid >> 5;
  const int rt = wave & 3, cg = wave >> 2;
  const int half = lane >> 4, lc = lane & 15;

  v8f acc[4] = {};
  const unsigned short* ap = As + (rt * 16 + lc) * AS_STRIDE;
#pragma unroll
  for (int k0 = 0; k0 < DCH; k0 += 32) {
    v16bf a = load_a_frag(ap, k0, half);
#pragma unroll
    for (int t = 0; t < 4; ++t) {
      int n = (cg * 4 + t) * 16 + lc;
      v16bf b = load_b_frag(Wt + n * WS_STRIDE + k0 + half * 8);
      acc[t] = __builtin_amdgcn_wmma_f32_16x16x32_bf16(
          false, a, false, b, (short)0, acc[t], false, false);
    }
  }

#pragma unroll
  for (int t = 0; t < 4; ++t) {
    const int col = (cg * 4 + t) * 16 + lc;
    const float bv = bs[col];
#pragma unroll
    for (int v = 0; v < 8; ++v) {
      int row = row0 + rt * 16 + half * 8 + v;
      if (row < N_NODES)
        hout[(size_t)row * DCH + col] = eluf(acc[t][v] + bv);
    }
  }
}

// ---- per-graph mean pool: scatter sums + counts ---------------------------
__global__ __launch_bounds__(256)
void pool_kernel(const float* __restrict__ h, const int* __restrict__ batch,
                 float* __restrict__ pool, float* __restrict__ cnt) {
  long t = (long)blockIdx.x * 256 + threadIdx.x;
  int node = (int)(t >> 5), lane = (int)(t & 31);
  if (node >= N_NODES) return;
  int g = batch[node];
  const float4 v = *(const float4*)(h + (size_t)node * DCH + lane * 4);
  float* p = pool + (size_t)g * DCH + lane * 4;
  gatomic_add(p + 0, v.x); gatomic_add(p + 1, v.y);
  gatomic_add(p + 2, v.z); gatomic_add(p + 3, v.w);
  if (lane == 0) gatomic_add(&cnt[g], 1.f);
}

// ---- head: out[g] = (pool[g]/max(cnt,1)) @ lin_W + lin_b ------------------
__global__ void head_kernel(const float* __restrict__ pool, const float* __restrict__ cnt,
                            const float* __restrict__ linW, const float* __restrict__ linb,
                            float* __restrict__ out) {
  int g = blockIdx.x * blockDim.x + threadIdx.x;
  if (g >= NGRAPH) return;
  float inv = 1.f / fmaxf(cnt[g], 1.f);
  float a0 = linb[0], a1 = linb[1];
#pragma unroll 4
  for (int k = 0; k < DCH; ++k) {
    float p = pool[(size_t)g * DCH + k] * inv;
    a0 += p * linW[2 * k];
    a1 += p * linW[2 * k + 1];
  }
  out[2 * g]     = a0;
  out[2 * g + 1] = a1;
}

// ---------------------------------------------------------------------------
extern "C" void kernel_launch(void* const* d_in, const int* in_sizes, int n_in,
                              void* d_out, int out_size, void* d_ws, size_t ws_size,
                              hipStream_t stream) {
  (void)in_sizes; (void)n_in; (void)out_size; (void)ws_size;

  const float* x     = (const float*)d_in[0];
  const int*   eidx  = (const int*)  d_in[1];
  const int*   batch = (const int*)  d_in[2];
  const float* W1    = (const float*)d_in[3];
  const float* b1    = (const float*)d_in[4];
  const float* gamma = (const float*)d_in[5];
  const float* beta  = (const float*)d_in[6];
  const float* W2    = (const float*)d_in[7];
  const float* b2    = (const float*)d_in[8];
  const float* linW  = (const float*)d_in[9];
  const float* linb  = (const float*)d_in[10];
  float* out = (float*)d_out;

  // Workspace layout (floats + trailing bf16 weights): ~103 MB total.
  float* ws    = (float*)d_ws;
  float* hbuf  = ws;                                  // N*D activations
  float* zbuf  = hbuf + (size_t)N_NODES * DCH;        // N*D: agg, then z (in place)
  float* gsum  = zbuf + (size_t)N_NODES * DCH;        // 128
  float* gsq   = gsum + DCH;                          // 128
  float* scale = gsq + DCH;                           // 128
  float* shift = scale + DCH;                         // 128
  float* pool  = shift + DCH;                         // G*D
  float* cnt   = pool + (size_t)NGRAPH * DCH;         // G
  unsigned short* wt1g = (unsigned short*)(cnt + NGRAPH);  // D*D bf16 (16B aligned)
  unsigned short* wt2g = wt1g + DCH * DCH;                 // D*D bf16

  const int* src = eidx;
  const int* dst = eidx + N_EDGES;

  const dim3 blk(256);
  const int nd4 = (N_NODES * DCH) / 4;                        // 3.2M float4
  const dim3 gZeroAgg((nd4 + 255) / 256);
  const dim3 gScatter((N_EDGES * 32) / 256);                  // 200000 blocks
  const dim3 gGemm((N_NODES + 63) / 64);                      // 1563 blocks
  const dim3 gPool((N_NODES * 32 + 255) / 256);               // 12500 blocks

  for (int l = 0; l < NLAYERS; ++l) {
    const float* hin = (l == 0) ? x : hbuf;
    prep_weights<<<dim3(32), blk, 0, stream>>>(W1 + (size_t)l * DCH * DCH,
                                               W2 + (size_t)l * DCH * DCH,
                                               wt1g, wt2g);
    zero4_kernel<<<gZeroAgg, blk, 0, stream>>>((float4*)zbuf, nd4);
    zero4_kernel<<<dim3(1), blk, 0, stream>>>((float4*)gsum, (2 * DCH) / 4);
    scatter_kernel<<<gScatter, blk, 0, stream>>>(hin, src, dst, zbuf);
    gin_gemm1<<<gGemm, blk, 0, stream>>>(hin, zbuf, wt1g, b1 + l * DCH,
                                         zbuf, gsum, gsq);
    bn_finalize<<<dim3(1), dim3(128), 0, stream>>>(gsum, gsq,
                                                   gamma + l * DCH, beta + l * DCH,
                                                   scale, shift);
    gin_gemm2<<<gGemm, blk, 0, stream>>>(zbuf, scale, shift, wt2g, b2 + l * DCH,
                                         hbuf);
  }

  zero4_kernel<<<dim3(((NGRAPH * DCH + NGRAPH) / 4 + 255) / 256), blk, 0, stream>>>(
      (float4*)pool, (NGRAPH * DCH + NGRAPH) / 4);
  pool_kernel<<<gPool, blk, 0, stream>>>(hbuf, batch, pool, cnt);
  head_kernel<<<dim3((NGRAPH + 255) / 256), blk, 0, stream>>>(pool, cnt, linW, linb, out);
}
